// GPS_37477884625104
// MI455X (gfx1250) — matrix operand order repaired and checked by reading, not compiled, not run on hardware
//
#include <hip/hip_runtime.h>
#include <stdint.h>

// ---------------- constants ----------------
#define N_NODES 32768
#define N_EDGES 524288
#define H_DIM   128
#define QKV_LD  384
#define S_LEN   512
#define BATCH   64
#define NHEAD   4

typedef _Float16 fp16x8  __attribute__((ext_vector_type(8)));
typedef _Float16 fp16x16 __attribute__((ext_vector_type(16)));
typedef float    fx8     __attribute__((ext_vector_type(8)));

// Load a 16-element f16 fragment from two contiguous 16B chunks.
__device__ __forceinline__ fp16x16 ld_frag(const _Float16* p0, const _Float16* p1) {
  fp16x8 lo = *(const fp16x8*)p0;
  fp16x8 hi = *(const fp16x8*)p1;
  return __builtin_shufflevector(lo, hi, 0,1,2,3,4,5,6,7,8,9,10,11,12,13,14,15);
}

__device__ __forceinline__ fx8 wmma16(fp16x16 a, fp16x16 b, fx8 c) {
  // D = A(16x32 f16) * B(32x16 f16) + C(16x16 f32)
  return __builtin_amdgcn_wmma_f32_16x16x32_f16(false, a, false, b, (short)0, c, false, false);
}

// ---------------- generic WMMA GEMM ----------------
// C[M,Cout] = act( A[M,K](f16) @ W[Cout,K](f16)^T + bias )
// grid.x = M/16, 8 waves/block; wave w handles column tiles w, w+8, ...
template<int K>
__global__ __launch_bounds__(256) void k_gemm(
    const _Float16* __restrict__ A, const _Float16* __restrict__ W,
    const float* __restrict__ bias, float* __restrict__ Cf,
    _Float16* __restrict__ Ch, int Cout, int dorelu)
{
  constexpr int NK = K / 32;
  constexpr int STR = K + 8;           // padded LDS stride (halves)
  __shared__ _Float16 sA[16 * STR];

  const int tid  = threadIdx.x;
  const int row0 = blockIdx.x * 16;

  // stage A tile (16 x K) into LDS via CDNA5 async memory->LDS DMA
  // (sA is the only LDS object in this kernel -> LDS offset 0)
  #pragma unroll
  for (int i = tid; i < (16 * K) / 8; i += 256) {
    const int idx = i * 8;
    const int r = idx / K, c = idx % K;
    const uint32_t lds_off = (uint32_t)(r * STR + c) * 2u;
    const _Float16* gp = &A[(size_t)(row0 + r) * K + c];
    asm volatile("global_load_async_to_lds_b128 %0, %1, off"
                 :: "v"(lds_off), "v"(gp) : "memory");
  }
  asm volatile("s_wait_asynccnt 0x0" ::: "memory");
  __syncthreads();

  const int lane = tid & 31;
  const int wave = __builtin_amdgcn_readfirstlane(tid >> 5);  // uniform -> scalar loops
  const int lh = lane >> 4, ln = lane & 15;

  // preload all A fragments for this wave's 16 rows
  fp16x16 af[NK];
  #pragma unroll
  for (int kt = 0; kt < NK; ++kt) {
    const _Float16* p = &sA[ln * STR + kt * 32 + 8 * lh];
    af[kt] = ld_frag(p, p + 16);
  }

  const int ntiles = Cout >> 4;
  for (int ct = wave; ct < ntiles; ct += 8) {
    const int col = ct * 16 + ln;
    const float bv = bias[col];
    fx8 acc = {bv, bv, bv, bv, bv, bv, bv, bv};
    #pragma unroll
    for (int kt = 0; kt < NK; ++kt) {
      const _Float16* bp = &W[(size_t)col * K + kt * 32 + 8 * lh];
      fp16x16 bf = ld_frag(bp, bp + 16);
      acc = wmma16(af[kt], bf, acc);
    }
    #pragma unroll
    for (int r = 0; r < 8; ++r) {
      float v = acc[r];
      if (dorelu) v = fmaxf(v, 0.f);
      const int row = row0 + r + 8 * lh;
      if (Cf) Cf[(size_t)row * Cout + col] = v;
      if (Ch) Ch[(size_t)row * Cout + col] = (_Float16)v;
    }
  }
}

// ---------------- attention (per batch,head,16-query tile) ----------------
// qkv: [N, 384] f16 (q=0..127, k=128..255, v=256..383), out: [N,128] f16
__global__ __launch_bounds__(256) void k_attn(
    const _Float16* __restrict__ qkv, _Float16* __restrict__ out)
{
  __shared__ __align__(16) unsigned char smem[91392];
  _Float16* sV   = (_Float16*)smem;             // Vt [32][520] f16   (33280 B)
  float*    sS   = (float*)(smem + 33280);      // scores [16][520]   (33280 B)
  _Float16* sP   = (_Float16*)(smem + 66560);   // probs  [16][520]   (16640 B)
  float*    sRed = (float*)(smem + 83200);      // partials [8][16][16] (8192 B)

  const int bid   = blockIdx.x;
  const int qt    = bid & 31;
  const int h     = (bid >> 5) & 3;
  const int b     = bid >> 7;
  const int node0 = b * S_LEN;
  const int qrow0 = node0 + qt * 16;
  const int tid   = threadIdx.x;
  const int lane  = tid & 31;
  const int wave  = __builtin_amdgcn_readfirstlane(tid >> 5); // uniform
  const int lh    = lane >> 4, ln = lane & 15;

  // stage V transposed: sV[d][key]
  for (int i = tid; i < S_LEN * 32; i += 256) {
    int key = i >> 5, d = i & 31;
    sV[d * 520 + key] = qkv[(size_t)(node0 + key) * QKV_LD + 256 + h * 32 + d];
  }
  __syncthreads();

  // scores = (Q K^T) * 1/sqrt(32): fixed 4 tiles per wave
  {
    const _Float16* qb = &qkv[(size_t)(qrow0 + ln) * QKV_LD + h * 32 + 8 * lh];
    fp16x16 aq = ld_frag(qb, qb + 16);
    #pragma unroll
    for (int j = 0; j < 4; ++j) {
      const int kt = wave + j * 8;
      const _Float16* kb = &qkv[(size_t)(node0 + kt * 16 + ln) * QKV_LD + 128 + h * 32 + 8 * lh];
      fp16x16 bk = ld_frag(kb, kb + 16);
      fx8 acc = {0.f, 0.f, 0.f, 0.f, 0.f, 0.f, 0.f, 0.f};
      acc = wmma16(aq, bk, acc);
      #pragma unroll
      for (int r = 0; r < 8; ++r)
        sS[(r + 8 * lh) * 520 + kt * 16 + ln] = acc[r] * 0.17677669529663687f;
    }
  }
  __syncthreads();

  // softmax over 512 keys: 16 threads per row, 32 cols each
  {
    const int row = tid >> 4, p = tid & 15;
    float* srow = &sS[row * 520];
    float mx = -1e30f;
    for (int c = p * 32; c < p * 32 + 32; ++c) mx = fmaxf(mx, srow[c]);
    #pragma unroll
    for (int o = 1; o < 16; o <<= 1) mx = fmaxf(mx, __shfl_xor(mx, o, 32));
    float sum = 0.f;
    for (int c = p * 32; c < p * 32 + 32; ++c) {
      float e = __expf(srow[c] - mx);
      srow[c] = e; sum += e;
    }
    #pragma unroll
    for (int o = 1; o < 16; o <<= 1) sum += __shfl_xor(sum, o, 32);
    const float inv = 1.f / sum;
    _Float16* prow = &sP[row * 520];
    for (int c = p * 32; c < p * 32 + 32; ++c) prow[c] = (_Float16)(srow[c] * inv);
  }
  __syncthreads();

  // O = P @ V : 2 column tiles (hd=32), contraction split over 4 wave-groups
  {
    const int nsub = wave & 1, ks = wave >> 1;  // ks in 0..3 (uniform)
    fx8 acc = {0.f, 0.f, 0.f, 0.f, 0.f, 0.f, 0.f, 0.f};
    #pragma unroll
    for (int j = 0; j < 4; ++j) {
      const int ck = ks * 4 + j;                // 32-key chunk
      const _Float16* pa = &sP[ln * 520 + ck * 32 + 8 * lh];
      fp16x16 ap = ld_frag(pa, pa + 16);
      const _Float16* vb = &sV[(nsub * 16 + ln) * 520 + ck * 32 + 8 * lh];
      fp16x16 bv = ld_frag(vb, vb + 16);
      acc = wmma16(ap, bv, acc);
    }
    #pragma unroll
    for (int r = 0; r < 8; ++r)
      sRed[wave * 256 + (r + 8 * lh) * 16 + ln] = acc[r];
  }
  __syncthreads();

  // reduce 4 partials and write [16 rows x 32 dims]
  for (int i = tid; i < 512; i += 256) {
    const int m = i >> 5, d = i & 31;
    const int ns = d >> 4, n = d & 15;
    float s = 0.f;
    #pragma unroll
    for (int k = 0; k < 4; ++k) s += sRed[(k * 2 + ns) * 256 + m * 16 + n];
    out[(size_t)(qrow0 + m) * H_DIM + h * 32 + d] = (_Float16)s;
  }
}

// ---------------- edge gather / scatter (GINE aggregation) ----------------
__global__ __launch_bounds__(256) void k_edge(
    const float* __restrict__ x, const int* __restrict__ ei,
    const float* __restrict__ ea, float* __restrict__ aggr)
{
  const int gid = blockIdx.x * 256 + threadIdx.x;
  const int e = gid >> 5;
  if (e >= N_EDGES) return;
  const int c = (gid & 31) * 4;
  const int src = ei[e], dst = ei[N_EDGES + e];
  const float4 xv = *(const float4*)&x[(size_t)src * H_DIM + c];
  const float4 ev = *(const float4*)&ea[(size_t)e * H_DIM + c];
  float* d = &aggr[(size_t)dst * H_DIM + c];
  atomicAdd(d + 0, fmaxf(xv.x + ev.x, 0.f));
  atomicAdd(d + 1, fmaxf(xv.y + ev.y, 0.f));
  atomicAdd(d + 2, fmaxf(xv.z + ev.z, 0.f));
  atomicAdd(d + 3, fmaxf(xv.w + ev.w, 0.f));
}

// ---------------- elementwise helpers ----------------
__global__ void k_zero(float* p, int n) {
  int i = blockIdx.x * 256 + threadIdx.x;
  if (i < n) p[i] = 0.f;
}
__global__ void k_f32_to_f16(const float* __restrict__ s, _Float16* __restrict__ d, int n) {
  int i = blockIdx.x * 256 + threadIdx.x;
  if (i < n) d[i] = (_Float16)s[i];
}
__global__ void k_add2h(const float* __restrict__ a, const float* __restrict__ b,
                        _Float16* __restrict__ d, int n) {
  int i = blockIdx.x * 256 + threadIdx.x;
  if (i < n) d[i] = (_Float16)(a[i] + b[i]);
}
#define BN_INV 0.9999950000374997f
// out = bn1(hl + x) + bn2(hg + x); write f32 + f16
__global__ void k_combine(const float* __restrict__ hl, const float* __restrict__ hg,
                          const float* __restrict__ x, const float* __restrict__ g,
                          const float* __restrict__ bb, float* __restrict__ o32,
                          _Float16* __restrict__ oh, int l) {
  int i = blockIdx.x * 256 + threadIdx.x;
  if (i >= N_NODES * H_DIM) return;
  int c = i & 127;
  float s1 = g[(l * 3 + 0) * H_DIM + c] * BN_INV, b1 = bb[(l * 3 + 0) * H_DIM + c];
  float s2 = g[(l * 3 + 1) * H_DIM + c] * BN_INV, b2 = bb[(l * 3 + 1) * H_DIM + c];
  float xv = x[i];
  float o = (hl[i] + xv) * s1 + b1 + (hg[i] + xv) * s2 + b2;
  o32[i] = o;
  oh[i] = (_Float16)o;
}
// x = bn3(o + m), optional relu; write f32 + f16
__global__ void k_bn3(const float* __restrict__ o, const float* __restrict__ m,
                      const float* __restrict__ g, const float* __restrict__ bb,
                      float* __restrict__ xf, _Float16* __restrict__ xh,
                      int l, int dorelu) {
  int i = blockIdx.x * 256 + threadIdx.x;
  if (i >= N_NODES * H_DIM) return;
  int c = i & 127;
  float s3 = g[(l * 3 + 2) * H_DIM + c] * BN_INV, b3 = bb[(l * 3 + 2) * H_DIM + c];
  float v = (o[i] + m[i]) * s3 + b3;
  if (dorelu) v = fmaxf(v, 0.f);
  xf[i] = v;
  xh[i] = (_Float16)v;
}

// ---------------- host orchestration ----------------
extern "C" void kernel_launch(void* const* d_in, const int* in_sizes, int n_in,
                              void* d_out, int out_size, void* d_ws, size_t ws_size,
                              hipStream_t stream) {
  const float* x_in      = (const float*)d_in[0];
  const int*   ei        = (const int*)d_in[1];
  const float* ea        = (const float*)d_in[3];
  const float* lin_in_w  = (const float*)d_in[4];
  const float* lin_in_b  = (const float*)d_in[5];
  const float* lin_out_w = (const float*)d_in[6];
  const float* lin_out_b = (const float*)d_in[7];
  const float* gine_w1   = (const float*)d_in[8];
  const float* gine_b1   = (const float*)d_in[9];
  const float* gine_w2   = (const float*)d_in[10];
  const float* gine_b2   = (const float*)d_in[11];
  const float* ain_w     = (const float*)d_in[12];
  const float* ain_b     = (const float*)d_in[13];
  const float* aout_w    = (const float*)d_in[14];
  const float* aout_b    = (const float*)d_in[15];
  const float* m1_w      = (const float*)d_in[16];
  const float* m1_b      = (const float*)d_in[17];
  const float* m2_w      = (const float*)d_in[18];
  const float* m2_b      = (const float*)d_in[19];
  const float* bn_g      = (const float*)d_in[20];
  const float* bn_b      = (const float*)d_in[21];

  uint8_t* w = (uint8_t*)d_ws;
  const size_t MB = 1024 * 1024;
  float*    xf32 = (float*)(w + 0);           // [N,128] f32
  _Float16* xh   = (_Float16*)(w + 16 * MB);  // [N,128] f16 (also 'oh')
  float*    aggr = (float*)(w + 24 * MB);     // [N,128] f32 (also 'out' o32)
  _Float16* t0h  = (_Float16*)(w + 40 * MB);  // [N,384] f16 qkv / [N,256] mlp hidden
  _Float16* h16a = (_Float16*)(w + 64 * MB);  // [N,128] f16 gine hidden / attn out
  _Float16* h16b = (_Float16*)(w + 72 * MB);  // [N,128] f16 (x+aggr) / input f16
  float*    t1a  = (float*)(w + 80 * MB);     // [N,128] f32 gine out
  float*    t1b  = (float*)(w + 96 * MB);     // [N,128] f32 attn out / mlp out
  _Float16* wh   = (_Float16*)(w + 112 * MB); // f16 weights (~1 MB)

  _Float16* wh_lin_in  = wh;
  _Float16* wh_lin_out = wh + 16384;
  _Float16* wh_layer   = wh + 24576;
  const int LSTRIDE = 163840; // per-layer f16 weight stride (halves)

  const int NHE = N_NODES * H_DIM;
  const dim3 blk(256);
  auto cdiv = [](int a, int b) { return (a + b - 1) / b; };

  // convert weights to f16 (cheap; stays hot in L2)
  k_f32_to_f16<<<cdiv(16384, 256), blk, 0, stream>>>(lin_in_w, wh_lin_in, 16384);
  k_f32_to_f16<<<cdiv(8192, 256), blk, 0, stream>>>(lin_out_w, wh_lin_out, 8192);
  for (int l = 0; l < 3; ++l) {
    _Float16* base = wh_layer + (size_t)l * LSTRIDE;
    k_f32_to_f16<<<64,  blk, 0, stream>>>(gine_w1 + (size_t)l * 16384, base + 0,      16384);
    k_f32_to_f16<<<64,  blk, 0, stream>>>(gine_w2 + (size_t)l * 16384, base + 16384,  16384);
    k_f32_to_f16<<<192, blk, 0, stream>>>(ain_w   + (size_t)l * 49152, base + 32768,  49152);
    k_f32_to_f16<<<64,  blk, 0, stream>>>(aout_w  + (size_t)l * 16384, base + 81920,  16384);
    k_f32_to_f16<<<128, blk, 0, stream>>>(m1_w    + (size_t)l * 32768, base + 98304,  32768);
    k_f32_to_f16<<<128, blk, 0, stream>>>(m2_w    + (size_t)l * 32768, base + 131072, 32768);
  }

  // lin_in: x = x_in @ W^T + b
  k_f32_to_f16<<<cdiv(NHE, 256), blk, 0, stream>>>(x_in, h16b, NHE);
  k_gemm<128><<<N_NODES / 16, blk, 0, stream>>>(h16b, wh_lin_in, lin_in_b, xf32, xh, 128, 0);

  for (int l = 0; l < 3; ++l) {
    _Float16* base = wh_layer + (size_t)l * LSTRIDE;

    // GINE aggregation: aggr[dst] += relu(x[src] + e)
    k_zero<<<cdiv(NHE, 256), blk, 0, stream>>>(aggr, NHE);
    k_edge<<<(N_EDGES * 32) / 256, blk, 0, stream>>>(xf32, ei, ea, aggr);
    k_add2h<<<cdiv(NHE, 256), blk, 0, stream>>>(xf32, aggr, h16b, NHE);

    // GINE MLP
    k_gemm<128><<<N_NODES / 16, blk, 0, stream>>>(h16b, base,         gine_b1 + l * 128, nullptr, h16a, 128, 1);
    k_gemm<128><<<N_NODES / 16, blk, 0, stream>>>(h16a, base + 16384, gine_b2 + l * 128, t1a, nullptr, 128, 0);

    // attention: qkv projection, MHA, output projection
    k_gemm<128><<<N_NODES / 16, blk, 0, stream>>>(xh, base + 32768, ain_b + l * 384, nullptr, t0h, 384, 0);
    k_attn<<<BATCH * NHEAD * (S_LEN / 16), blk, 0, stream>>>(t0h, h16a);
    k_gemm<128><<<N_NODES / 16, blk, 0, stream>>>(h16a, base + 81920, aout_b + l * 128, t1b, nullptr, 128, 0);

    // out = bn1(h+x) + bn2(ha+x)
    k_combine<<<cdiv(NHE, 256), blk, 0, stream>>>(t1a, t1b, xf32, bn_g, bn_b, aggr, xh, l);

    // FFN
    k_gemm<128><<<N_NODES / 16, blk, 0, stream>>>(xh, base + 98304,  m1_b + l * 256, nullptr, t0h, 256, 1);
    k_gemm<256><<<N_NODES / 16, blk, 0, stream>>>(t0h, base + 131072, m2_b + l * 128, t1b, nullptr, 128, 0);

    // x = bn3(out + m) (+relu for l<2)
    k_bn3<<<cdiv(NHE, 256), blk, 0, stream>>>(aggr, t1b, bn_g, bn_b, xf32, xh, l, (l < 2) ? 1 : 0);
  }

  // lin_out -> d_out [N,64] f32
  k_gemm<128><<<N_NODES / 16, blk, 0, stream>>>(xh, wh_lin_out, lin_out_b, (float*)d_out, nullptr, 64, 0);
}
